// GCN3_27410481283082
// MI455X (gfx1250) — matrix operand order repaired and checked by reading, not compiled, not run on hardware
//
#include <hip/hip_runtime.h>
#include <hip/hip_bf16.h>

typedef _Float16 half_t;
typedef __attribute__((ext_vector_type(8)))  _Float16 v8h;
typedef __attribute__((ext_vector_type(16))) _Float16 v16h;
typedef __attribute__((ext_vector_type(8)))  float    v8f;
typedef __attribute__((ext_vector_type(4)))  unsigned int tdm_u32x4;
typedef __attribute__((ext_vector_type(8)))  int          tdm_i32x8;
typedef __attribute__((ext_vector_type(4)))  int          tdm_i32x4;

#if __has_builtin(__builtin_amdgcn_tensor_load_to_lds) && __has_builtin(__builtin_amdgcn_s_wait_tensorcnt)
#define HAVE_TDM 1
#pragma message("CDNA5 probe: __builtin_amdgcn_tensor_load_to_lds AVAILABLE")
#else
#define HAVE_TDM 0
#pragma message("CDNA5 probe: __builtin_amdgcn_tensor_load_to_lds NOT available")
#endif
#if __has_include(<hip/amd_detail/amd_gfx1250_TDM.h>)
#define TDM_SIX_ARGS 1
#pragma message("CDNA5 probe: TDM header present -> 6-arg form")
#else
#define TDM_SIX_ARGS 0
#pragma message("CDNA5 probe: no TDM header -> 5-arg form")
#endif

// ---------------------------------------------------------------- utilities

__global__ void k_zero_f32(float* __restrict__ p, int n) {
  int i = blockIdx.x * blockDim.x + threadIdx.x;
  if (i < n) p[i] = 0.f;
}

__global__ void k_f32_to_f16(const float* __restrict__ s, half_t* __restrict__ d, int n) {
  int i = blockIdx.x * blockDim.x + threadIdx.x;
  if (i < n) d[i] = (half_t)s[i];
}

__global__ void k_f16_to_f32(const half_t* __restrict__ s, float* __restrict__ d, int n) {
  int i = blockIdx.x * blockDim.x + threadIdx.x;
  if (i < n) d[i] = (float)s[i];
}

// ------------------------------------------------- implicit-GEMM WMMA conv
// 3x3 SAME conv as GEMM: M = OC (16-tile), N = TWxTH spatial tile (16/wave),
// K = IC*9 padded to 32.  Weights staged to LDS via TDM tensor_load_to_lds
// when K%32==0 (no LDS pad needed), else cooperative loads.  Input halo tile
// staged to LDS.  A-fragments are two ds_load_b128; B-fragment gathers use
// compile-time immediate LDS offsets (grp split by a divergent branch so the
// k->(ic,kh,kw) decomposition constant-folds).  WMMA executes at full EXEC.
template<int IC, int OC, int H, int W>
__global__ __launch_bounds__(128) void k_conv_wmma(const half_t* __restrict__ in,
                                                   const half_t* __restrict__ wgt,
                                                   half_t* __restrict__ out) {
  constexpr int K   = IC * 9;
  constexpr int KP  = (K + 31) & ~31;
  constexpr int TW  = (W >= 8) ? 8 : W;
  constexpr int TH  = ((64 / TW) > H) ? H : (64 / TW);
  constexpr int RPW = 16 / TW;                 // tile rows per wave
  constexpr int TILES_X = W / TW;
  constexpr int SH = TH + 2, SWD = TW + 2;     // staged tile with halo
  constexpr bool TDM_OK = HAVE_TDM && (K % 32 == 0);

  __shared__ half_t sW[16 * KP];
  __shared__ half_t sIn[IC * SH * SWD];

  const int img = blockIdx.x;
  const int ocb = blockIdx.z * 16;
  const int ty0 = (blockIdx.y / TILES_X) * TH;
  const int tx0 = (blockIdx.y % TILES_X) * TW;

  // ---- weight tile -> LDS
#if HAVE_TDM
  if constexpr (TDM_OK) {
    if (threadIdx.x < 32) {          // wave-uniform branch: only wave 0 issues TDM
      unsigned long long ga = (unsigned long long)(uintptr_t)(wgt + (size_t)ocb * K);
      unsigned int lds = (unsigned int)(uintptr_t)(&sW[0]);
      tdm_u32x4 g0; tdm_i32x8 g1;
      tdm_i32x4 g2 = {0, 0, 0, 0}, g3 = {0, 0, 0, 0};
      g0.x = 1u;                                        // count=1, user descriptor
      g0.y = lds;                                       // lds_addr
      g0.z = (unsigned int)ga;                          // global_addr[31:0]
      g0.w = (unsigned int)(ga >> 32) | (2u << 30);     // global_addr[56:32] | type=2
      g1[0] = (int)(1u << 16);                          // data_size=1 (2 bytes)
      g1[1] = (int)(((unsigned)K & 0xffffu) << 16);     // tensor_dim0[15:0]
      g1[2] = (int)(((unsigned)K >> 16) | (16u << 16)); // dim0[31:16] | tensor_dim1[15:0]=16
      g1[3] = (int)((unsigned)K << 16);                 // dim1[31:16]=0 | tile_dim0=K
      g1[4] = 16;                                       // tile_dim1=16 rows, tile_dim2=0
      g1[5] = K;                                        // tensor_dim0_stride[31:0]
      g1[6] = 0; g1[7] = 0;
#if TDM_SIX_ARGS
      tdm_i32x8 g4 = {0, 0, 0, 0, 0, 0, 0, 0};
      __builtin_amdgcn_tensor_load_to_lds(g0, g1, g2, g3, g4, 0);
#else
      __builtin_amdgcn_tensor_load_to_lds(g0, g1, g2, g3, 0);
#endif
      __builtin_amdgcn_s_wait_tensorcnt(0);
    }
  } else
#endif
  {
    for (int idx = threadIdx.x; idx < 16 * KP; idx += 128) {
      int m = idx / KP, k = idx % KP;
      sW[idx] = (k < K) ? wgt[(ocb + m) * K + k] : (half_t)0.f;
    }
  }

  // ---- input halo tile -> LDS (coalesced cooperative stage, zero-padded)
  for (int idx = threadIdx.x; idx < IC * SH * SWD; idx += 128) {
    int ic = idx / (SH * SWD), r = idx % (SH * SWD);
    int ry = r / SWD, rx = r % SWD;
    int iy = ty0 + ry - 1, ix = tx0 + rx - 1;
    half_t v = (half_t)0.f;
    if (iy >= 0 && iy < H && ix >= 0 && ix < W)
      v = in[((size_t)img * IC + ic) * (H * W) + iy * W + ix];
    sIn[idx] = v;
  }
  __syncthreads();

  const int lane = threadIdx.x & 31;
  const int wv   = threadIdx.x >> 5;
  const int grp  = lane >> 4;            // K-half selector: k = 16*grp + i (this code's contiguous order)
  const int nloc = lane & 15;            // N within tile / M row for A frag
  const int ly   = wv * RPW + nloc / TW; // tile-local output row
  const int lx   = nloc % TW;
  const bool pvalid = ly < TH;
  const int  lys = pvalid ? ly : 0;      // keep LDS reads in-bounds for idle waves

  // A base folds the grp*16 K-offset; the 16-element run is contiguous.
  const half_t* aBase = sW + nloc * KP + grp * 16;
  const half_t* bBase = sIn + lys * SWD + lx;

  v8f c = {};
  #pragma unroll
  for (int kb = 0; kb < KP; kb += 32) {
    // A frag: two aligned 16-byte LDS loads (K=kb+16g..+7, K=kb+16g+8..+15)
    const v8h* ap  = (const v8h*)(aBase + kb);
    v8h alo = ap[0];
    v8h ahi = ap[1];
    v16h a = __builtin_shufflevector(alo, ahi, 0, 1, 2, 3, 4, 5, 6, 7,
                                     8, 9, 10, 11, 12, 13, 14, 15);
    // B frag: lanes 0-15 K=kb..kb+15, lanes 16-31 K=kb+16..kb+31, N=lane&15.
    // Split on K-half so every (ic,kh,kw) decomposition is a compile-time
    // constant -> ds_load_u16 with immediate offsets, no per-element VALU.
    v16h b;
    if (lane < 16) {
      #pragma unroll
      for (int i = 0; i < 16; ++i) {
        constexpr int dummy = 0; (void)dummy;
        int k = kb + i;
        b[i] = (k < K) ? bBase[((k / 9) * SH + (k % 9) / 3) * SWD + (k % 9) % 3]
                       : (half_t)0.f;
      }
    } else {
      #pragma unroll
      for (int i = 0; i < 16; ++i) {
        int k = kb + 16 + i;
        b[i] = (k < K) ? bBase[((k / 9) * SH + (k % 9) / 3) * SWD + (k % 9) % 3]
                       : (half_t)0.f;
      }
    }
    c = __builtin_amdgcn_wmma_f32_16x16x32_f16(false, a, false, b,
                                               (short)0, c, false, false);
  }

  // C layout: VGPR r -> M = r + 8*grp, N = lane&15
  if (pvalid) {
    const int y = ty0 + ly, x = tx0 + lx;
    half_t* o = out + ((size_t)img * OC + ocb + grp * 8) * (H * W) + y * W + x;
    #pragma unroll
    for (int r = 0; r < 8; ++r) o[(size_t)r * (H * W)] = (half_t)c[r];
  }
}

// -------------------------------------------- BN batch stats (train mode)

__global__ void k_stats(const half_t* __restrict__ x, float* __restrict__ sums,
                        float* __restrict__ sumsq, int C, int HW) {
  __shared__ float ss[256], sq[256];
  const int plane = blockIdx.x;           // n*C + c
  const int c = plane % C;
  const int start = blockIdx.y * 1024;
  float s = 0.f, q = 0.f;
  for (int i = start + threadIdx.x; i < HW && i < start + 1024; i += 256) {
    float v = (float)x[(size_t)plane * HW + i];
    s += v; q += v * v;
  }
  ss[threadIdx.x] = s; sq[threadIdx.x] = q;
  __syncthreads();
  for (int st = 128; st > 0; st >>= 1) {
    if (threadIdx.x < st) { ss[threadIdx.x] += ss[threadIdx.x + st];
                            sq[threadIdx.x] += sq[threadIdx.x + st]; }
    __syncthreads();
  }
  if (threadIdx.x == 0) { atomicAdd(&sums[c], ss[0]); atomicAdd(&sumsq[c], sq[0]); }
}

__global__ void k_bn_finalize(const float* __restrict__ sums, const float* __restrict__ sumsq,
                              const float* __restrict__ g, const float* __restrict__ b,
                              float* __restrict__ scale, float* __restrict__ shift,
                              int C, float invCnt) {
  int c = threadIdx.x;
  if (c < C) {
    float m = sums[c] * invCnt;
    float v = sumsq[c] * invCnt - m * m;
    float r = rsqrtf(v + 1e-5f);
    float sc = g[c] * r;
    scale[c] = sc;
    shift[c] = b[c] - m * sc;
  }
}

__global__ void k_bn_relu_pool(const half_t* __restrict__ x, const float* __restrict__ scale,
                               const float* __restrict__ shift, half_t* __restrict__ out,
                               int C, int H, int W, int total) {
  int i = blockIdx.x * blockDim.x + threadIdx.x;
  if (i >= total) return;
  const int HO = H / 2, WO = W / 2;
  int px = i % WO, t = i / WO;
  int py = t % HO; t /= HO;
  int c = t % C;   int n = t / C;
  const half_t* src = x + ((size_t)n * C + c) * H * W;
  float sc = scale[c], sh = shift[c];
  float m = 0.f;
  #pragma unroll
  for (int dy = 0; dy < 2; ++dy)
    #pragma unroll
    for (int dx = 0; dx < 2; ++dx) {
      float v = (float)src[(2 * py + dy) * W + (2 * px + dx)] * sc + sh;
      v = fmaxf(v, 0.f);
      m = fmaxf(m, v);
    }
  out[i] = (half_t)m;
}

// ------------------------------------------------------------ graph head

__global__ void k_edge_cos(const float* __restrict__ feat, const int* __restrict__ src,
                           const int* __restrict__ dst, int nE,
                           float* __restrict__ nodeSum, float* __restrict__ nodeCnt) {
  int e = blockIdx.x * blockDim.x + threadIdx.x;
  if (e >= nE) return;
  const float* a = feat + (size_t)src[e] * 256;
  const float* b = feat + (size_t)dst[e] * 256;
  float d = 0.f, na = 0.f, nb = 0.f;
  for (int i = 0; i < 256; ++i) { d += a[i] * b[i]; na += a[i] * a[i]; nb += b[i] * b[i]; }
  float cosv = d / fmaxf(sqrtf(na) * sqrtf(nb), 1e-6f);
  atomicAdd(&nodeSum[dst[e]], cosv);
  atomicAdd(&nodeCnt[dst[e]], 1.f);
}

__global__ void k_node_mean(const float* __restrict__ s, const float* __restrict__ c,
                            float* __restrict__ o, int n) {
  int i = blockIdx.x * blockDim.x + threadIdx.x;
  if (i < n) o[i] = s[i] / fmaxf(c[i], 1.f);
}

// one block per graph, 256 threads = feature dims; 4 nodes per graph
__global__ void k_graph_pool(const float* __restrict__ feat, float* __restrict__ out) {
  int g = blockIdx.x, d = threadIdx.x;
  float s = 0.f;
  for (int j = 0; j < 4; ++j) s += feat[(size_t)(4 * g + j) * 256 + d];
  out[g * 256 + d] = s * 0.25f;
}

__global__ void k_pd1(const float* __restrict__ pcat, const float* __restrict__ scat,
                      float* __restrict__ pd1) {
  int i = blockIdx.x * blockDim.x + threadIdx.x;
  if (i >= 24 * 48) return;
  int q = i / 48, s = i % 48;
  float acc = 0.f;
  for (int d = 0; d < 256; ++d) {
    float df = pcat[q * 256 + d] - scat[s * 256 + d];
    acc += df * df;
  }
  pd1[i] = expf(-acc);
}

__global__ void k_class_mean(const float* __restrict__ pd1, const int* __restrict__ sy,
                             float* __restrict__ outm) {
  int i = blockIdx.x * blockDim.x + threadIdx.x;
  if (i >= 24 * 24) return;
  int q = i / 24, c = i % 24;
  float s = 0.f, cnt = 0.f;
  for (int g = 0; g < 48; ++g)
    if (sy[g] == c) { s += pd1[q * 48 + g]; cnt += 1.f; }
  outm[i] = s / fmaxf(cnt, 1.f);
}

__global__ void k_loss(const float* __restrict__ nodeCos, const int* __restrict__ sbatch,
                       int Ns, const float* __restrict__ outm, const int* __restrict__ qy,
                       float* __restrict__ res) {
  __shared__ float gsum[48], gcnt[48], sq[48], ce[24], corr[24];
  int t = threadIdx.x;
  if (t < 48) { gsum[t] = 0.f; gcnt[t] = 0.f; }
  __syncthreads();
  if (t < Ns) { int g = sbatch[t]; atomicAdd(&gsum[g], nodeCos[t]); atomicAdd(&gcnt[g], 1.f); }
  __syncthreads();
  if (t < 48) { float gm = gsum[t] / fmaxf(gcnt[t], 1.f); float d = gm - 1.f; sq[t] = d * d; }
  if (t < 24) {
    const float* row = outm + t * 24;
    float mx = row[0]; int am = 0;
    for (int c = 1; c < 24; ++c) if (row[c] > mx) { mx = row[c]; am = c; }
    float lse = 0.f;
    for (int c = 0; c < 24; ++c) lse += expf(row[c] - mx);
    lse = logf(lse) + mx;
    int lbl = qy[t];
    ce[t]   = lse - row[lbl];
    corr[t] = (am == lbl) ? 1.f : 0.f;
  }
  __syncthreads();
  if (t == 0) {
    float cl = 0.f;
    for (int i = 0; i < 48; ++i) cl += sq[i];
    float tot = 0.f, acc = 0.f;
    for (int q = 0; q < 24; ++q) { tot += ce[q]; acc += corr[q]; }
    float att  = ce[5];                       // ATT_CLASS row
    float rest = (tot - att) / 23.f;
    res[0] = att + rest + cl;  // loss
    res[1] = acc;              // acc
    res[2] = cl;               // cos_loss
    res[3] = att;              // atten_loss
    res[4] = rest;             // rest_loss
  }
}

// ---------------------------------------------------------------- launcher

static inline int cdiv(int a, int b) { return (a + b - 1) / b; }
static inline int conv_tiles(int H, int W) {
  int tw = (W >= 8) ? 8 : W;
  int th = (64 / tw > H) ? H : (64 / tw);
  return (H / th) * (W / tw);
}

extern "C" void kernel_launch(void* const* d_in, const int* in_sizes, int n_in,
                              void* d_out, int out_size, void* d_ws, size_t ws_size,
                              hipStream_t stream) {
  (void)n_in; (void)out_size; (void)ws_size;
  const float* support_x = (const float*)d_in[0];
  const float* query_x   = (const float*)d_in[1];
  const int*   s_ei      = (const int*)d_in[2];
  const int*   s_batch   = (const int*)d_in[4];
  const int*   s_y       = (const int*)d_in[6];
  const int*   q_y       = (const int*)d_in[7];
  const float* w1 = (const float*)d_in[8];
  const float* g1 = (const float*)d_in[9];
  const float* b1 = (const float*)d_in[10];
  const float* w2 = (const float*)d_in[11];
  const float* g2 = (const float*)d_in[12];
  const float* b2 = (const float*)d_in[13];
  const float* wr = (const float*)d_in[14];
  const float* gr = (const float*)d_in[15];
  const float* br = (const float*)d_in[16];
  float* out = (float*)d_out;

  // ---- workspace carving
  char* ws = (char*)d_ws;
  size_t cur = 0;
  auto carve = [&](size_t bytes) -> char* {
    char* p = ws + cur;
    cur += (bytes + 255) & ~(size_t)255;
    return p;
  };
  half_t* w16     = (half_t*)carve(166752 * sizeof(half_t));          // all weights f16
  float*  stats   = (float*)carve(4 * 64 * sizeof(float));            // sum|sumsq|scale|shift
  float*  feat_s  = (float*)carve(192 * 256 * sizeof(float));
  float*  feat_q  = (float*)carve(96 * 256 * sizeof(float));
  float*  nodeSum = (float*)carve(192 * sizeof(float));
  float*  nodeCnt = (float*)carve(192 * sizeof(float));
  float*  nodeCos = (float*)carve(192 * sizeof(float));
  float*  scat    = (float*)carve(48 * 256 * sizeof(float));
  float*  pcat    = (float*)carve(24 * 256 * sizeof(float));
  float*  pd1     = (float*)carve(24 * 48 * sizeof(float));
  float*  outm    = (float*)carve(24 * 24 * sizeof(float));
  half_t* bufX    = (half_t*)carve((size_t)52 * 1024 * 1024);
  half_t* bufY    = (half_t*)carve((size_t)52 * 1024 * 1024);
  half_t* bufC    = (half_t*)carve((size_t)202 * 1024 * 1024);

  float* sums  = stats;        float* sumsq = stats + 64;
  float* scale = stats + 128;  float* shift = stats + 192;

  // ---- weights -> f16 (OIHW flatten == [OC][IC*9] GEMM-A layout)
  const size_t wo[6] = {0, 864, 19296, 56160, 93024, 129888};
  k_f32_to_f16<<<cdiv(864, 256), 256, 0, stream>>>(w1, w16 + wo[0], 864);
  k_f32_to_f16<<<cdiv(18432, 256), 256, 0, stream>>>(w2, w16 + wo[1], 18432);
  k_f32_to_f16<<<cdiv(147456, 256), 256, 0, stream>>>(wr, w16 + wo[2], 147456);

  const int    OCt[6] = {32, 64, 64, 64, 64, 64};
  const int    Hs[6]  = {128, 64, 32, 16, 8, 4};
  const float* gms[6] = {g1, g2, gr, gr + 64, gr + 128, gr + 192};
  const float* bts[6] = {b1, b2, br, br + 64, br + 128, br + 192};

  for (int ph = 0; ph < 2; ++ph) {
    const int    N    = (ph == 0) ? 192 : 96;
    const float* xin  = (ph == 0) ? support_x : query_x;
    float*       feat = (ph == 0) ? feat_s : feat_q;

    const int nIn = N * 3 * 128 * 128;
    k_f32_to_f16<<<cdiv(nIn, 256), 256, 0, stream>>>(xin, bufX, nIn);

    half_t* src = bufX;
    half_t* dst = bufY;
    for (int blk = 0; blk < 6; ++blk) {
      const int H = Hs[blk], W = H, HW = H * W, C = OCt[blk];
      const half_t* wg = w16 + wo[blk];
      dim3 cg(N, conv_tiles(H, W), C / 16);
      switch (blk) {
        case 0: k_conv_wmma<3, 32, 128, 128><<<cg, 128, 0, stream>>>(src, wg, bufC); break;
        case 1: k_conv_wmma<32, 64, 64, 64><<<cg, 128, 0, stream>>>(src, wg, bufC); break;
        case 2: k_conv_wmma<64, 64, 32, 32><<<cg, 128, 0, stream>>>(src, wg, bufC); break;
        case 3: k_conv_wmma<64, 64, 16, 16><<<cg, 128, 0, stream>>>(src, wg, bufC); break;
        case 4: k_conv_wmma<64, 64, 8, 8><<<cg, 128, 0, stream>>>(src, wg, bufC); break;
        default: k_conv_wmma<64, 64, 4, 4><<<cg, 128, 0, stream>>>(src, wg, bufC); break;
      }
      k_zero_f32<<<1, 128, 0, stream>>>(sums, 128);           // sums + sumsq
      dim3 sg(N * C, cdiv(HW, 1024));
      k_stats<<<sg, 256, 0, stream>>>(bufC, sums, sumsq, C, HW);
      k_bn_finalize<<<1, 64, 0, stream>>>(sums, sumsq, gms[blk], bts[blk],
                                          scale, shift, C, 1.f / (float)(N * HW));
      const int tot = N * C * (H / 2) * (W / 2);
      k_bn_relu_pool<<<cdiv(tot, 256), 256, 0, stream>>>(bufC, scale, shift, dst, C, H, W, tot);
      half_t* tmp = src; src = dst; dst = tmp;
    }
    // final pooled [N][64][2][2] f16 == [N][256] features
    k_f16_to_f32<<<cdiv(N * 256, 256), 256, 0, stream>>>(src, feat, N * 256);
  }

  // ---- graph head (support edges only; reference discards query aggregation)
  const int nEs = in_sizes[2] / 2;     // 768
  k_zero_f32<<<2, 256, 0, stream>>>(nodeSum, 192);
  k_zero_f32<<<2, 256, 0, stream>>>(nodeCnt, 192);
  k_edge_cos<<<cdiv(nEs, 128), 128, 0, stream>>>(feat_s, s_ei, s_ei + nEs, nEs, nodeSum, nodeCnt);
  k_node_mean<<<1, 192, 0, stream>>>(nodeSum, nodeCnt, nodeCos, 192);
  k_graph_pool<<<48, 256, 0, stream>>>(feat_s, scat);
  k_graph_pool<<<24, 256, 0, stream>>>(feat_q, pcat);
  k_pd1<<<cdiv(24 * 48, 128), 128, 0, stream>>>(pcat, scat, pd1);
  k_class_mean<<<cdiv(24 * 24, 128), 128, 0, stream>>>(pd1, s_y, outm);
  k_loss<<<1, 256, 0, stream>>>(nodeCos, s_batch, 192, outm, q_y, out);
}
